// SparseLinear_42193758716222
// MI455X (gfx1250) — matrix operand order repaired and checked by reading, not compiled, not run on hardware
//
#include <hip/hip_runtime.h>

// SparseLinear as dense f16-WMMA GEMM: out = x @ W^T + bias
// x: [8192, 4096] fp32, W: [4096, 4096] fp32 (pre-masked), bias: [4096] fp32
// Compute-bound -> downconvert to f16 on the fly, accumulate fp32 in WMMA.
// Wave tile 64x64 (4x4 WMMA tiles) to get 1.0 ds_load_b128 per v_wmma.

typedef _Float16 v4h  __attribute__((ext_vector_type(4)));
typedef _Float16 v8h  __attribute__((ext_vector_type(8)));
typedef _Float16 v16h __attribute__((ext_vector_type(16)));
typedef float    v8f  __attribute__((ext_vector_type(8)));

#define GEMM_M 8192
#define GEMM_N 4096
#define GEMM_K 4096
#define BM 256
#define BN 128
#define BK 32
#define LDK 40   // padded row stride in halves: 80 bytes, 16B-aligned, bank-conflict-free

__global__ __launch_bounds__(256)
void SparseLinear_wmma_kernel(const float* __restrict__ X,
                              const float* __restrict__ W,
                              const float* __restrict__ bias,
                              float* __restrict__ out)
{
    __shared__ _Float16 As[2][BM][LDK];   // 2*256*40*2 = 40960 B
    __shared__ _Float16 Bs[2][BN][LDK];   // 2*128*40*2 = 20480 B

    const int tid  = threadIdx.x;
    const int lane = tid & 31;
    const int wave = tid >> 5;            // 0..7
    const int waveM = (wave >> 1) * 64;   // 0,64,128,192
    const int waveN = (wave & 1) * 64;    // 0,64

    const int m0 = blockIdx.y * BM;
    const int n0 = blockIdx.x * BN;

    // Cooperative staging: 256 threads.
    // A tile 256x32 fp32 -> 8 float4/thread ; B tile 128x32 fp32 -> 4 float4/thread
    const int lr = tid >> 3;              // row 0..31 (per pass)
    const int lc = (tid & 7) * 4;         // col {0,4,...,28}

    v8f c[4][4];
#pragma unroll
    for (int i = 0; i < 4; ++i)
#pragma unroll
        for (int j = 0; j < 4; ++j)
            c[i][j] = (v8f)0.0f;

    float4 ar[8], br[4];

    auto gload = [&](int kc) {
#pragma unroll
        for (int p = 0; p < 8; ++p) {
            const int r = lr + p * 32;
            ar[p] = *(const float4*)(X + (size_t)(m0 + r) * GEMM_K + kc + lc);
        }
#pragma unroll
        for (int p = 0; p < 4; ++p) {
            const int r = lr + p * 32;
            br[p] = *(const float4*)(W + (size_t)(n0 + r) * GEMM_K + kc + lc);
        }
    };
    auto sstore = [&](int buf) {
#pragma unroll
        for (int p = 0; p < 8; ++p) {
            const int r = lr + p * 32;
            v4h a = { (_Float16)ar[p].x, (_Float16)ar[p].y,
                      (_Float16)ar[p].z, (_Float16)ar[p].w };
            *(v4h*)&As[buf][r][lc] = a;
        }
#pragma unroll
        for (int p = 0; p < 4; ++p) {
            const int r = lr + p * 32;
            v4h b = { (_Float16)br[p].x, (_Float16)br[p].y,
                      (_Float16)br[p].z, (_Float16)br[p].w };
            *(v4h*)&Bs[buf][r][lc] = b;
        }
    };

    gload(0);
    sstore(0);

    const int half = lane >> 4;   // 0 or 1
    const int l15  = lane & 15;

    // A fragment (16x32 f16): lanes<16 hold K {0..7,16..23}, lanes>=16 hold {8..15,24..31}
    const int akb = half ? 8 : 0;
    // B fragment (32x16 f16): lanes<16 hold K {0..7,16..23}, lanes>=16 hold {8..15,24..31}
    const int bkb = half ? 8 : 0;

    const int aRow0 = waveM + l15;        // + mt*16 in loop
    const int bRow0 = waveN + l15;        // + nt*16 in loop

    constexpr int KCHUNKS = GEMM_K / BK;  // 128
    for (int kc = 0; kc < KCHUNKS; ++kc) {
        __syncthreads();
        const int buf = kc & 1;

        if (kc + 1 < KCHUNKS) gload((kc + 1) * BK);

        v16h a[4];
#pragma unroll
        for (int mt = 0; mt < 4; ++mt) {
            const _Float16* p = &As[buf][aRow0 + mt * 16][0];
            ((v8h*)&a[mt])[0] = *(const v8h*)(p + akb);
            ((v8h*)&a[mt])[1] = *(const v8h*)(p + akb + 16);
        }
#pragma unroll
        for (int nt = 0; nt < 4; ++nt) {
            v16h b;
            const _Float16* p = &Bs[buf][bRow0 + nt * 16][0];
            ((v8h*)&b)[0] = *(const v8h*)(p + bkb);
            ((v8h*)&b)[1] = *(const v8h*)(p + bkb + 16);
#pragma unroll
            for (int mt = 0; mt < 4; ++mt)
                c[mt][nt] = __builtin_amdgcn_wmma_f32_16x16x32_f16(
                    /*neg_a=*/false, a[mt], /*neg_b=*/false, b,
                    /*c_mod=*/(short)0, c[mt][nt],
                    /*reuse_a=*/false, /*reuse_b=*/false);
        }

        if (kc + 1 < KCHUNKS) sstore(buf ^ 1);
    }

    // Epilogue: C/D layout -> VGPR r holds M = r + 8*half (lane split), N = lane&15.
#pragma unroll
    for (int nt = 0; nt < 4; ++nt) {
        const int col = n0 + waveN + nt * 16 + l15;
        const float bv = bias[col];
#pragma unroll
        for (int mt = 0; mt < 4; ++mt) {
            const int rbase = m0 + waveM + mt * 16 + half * 8;
#pragma unroll
            for (int r = 0; r < 8; ++r) {
                out[(size_t)(rbase + r) * GEMM_N + col] = c[mt][nt][r] + bv;
            }
        }
    }
}

extern "C" void kernel_launch(void* const* d_in, const int* in_sizes, int n_in,
                              void* d_out, int out_size, void* d_ws, size_t ws_size,
                              hipStream_t stream) {
    (void)in_sizes; (void)n_in; (void)out_size; (void)d_ws; (void)ws_size;
    const float* x    = (const float*)d_in[0];
    const float* w    = (const float*)d_in[1];  // already masked
    const float* bias = (const float*)d_in[2];
    // d_in[3] (mask) intentionally unused: weight is pre-masked in setup.
    float* out = (float*)d_out;

    dim3 grid(GEMM_N / BN, GEMM_M / BM);  // (32, 32)
    SparseLinear_wmma_kernel<<<grid, 256, 0, stream>>>(x, w, bias, out);
}
